// AdversarialWindowLoss_52733608460400
// MI455X (gfx1250) — matrix operand order repaired and checked by reading, not compiled, not run on hardware
//
#include <hip/hip_runtime.h>
#include <hip/hip_bf16.h>
#include <math.h>

typedef __attribute__((ext_vector_type(16))) _Float16 v16h;
typedef __attribute__((ext_vector_type(8)))  _Float16 v8h;
typedef __attribute__((ext_vector_type(8)))  float    v8f;

#define HH      768
#define WWID    768
#define HWSZ    (768*768)
#define BATCH   16
#define PATCH   64
#define NPP     4
#define NPATCH  64      // BATCH * NPP
#define TOPK_K  100

// ---------------------------------------------------------------------------
// Global min reduction (for the data-dependent [-1,1]->[0,1] remap)
// ---------------------------------------------------------------------------
__global__ void k_min_partial(const float* __restrict__ x, long n, float* __restrict__ partial) {
    __shared__ float red[256];
    float m = 3.4e38f;
    for (long i = (long)blockIdx.x * 256 + threadIdx.x; i < n; i += (long)gridDim.x * 256)
        m = fminf(m, x[i]);
    red[threadIdx.x] = m; __syncthreads();
    for (int s = 128; s > 0; s >>= 1) {
        if ((int)threadIdx.x < s) red[threadIdx.x] = fminf(red[threadIdx.x], red[threadIdx.x + s]);
        __syncthreads();
    }
    if (threadIdx.x == 0) partial[blockIdx.x] = red[0];
}

__global__ void k_min_final(const float* __restrict__ partial, int n, float* __restrict__ out) {
    __shared__ float red[256];
    float m = 3.4e38f;
    for (int i = threadIdx.x; i < n; i += 256) m = fminf(m, partial[i]);
    red[threadIdx.x] = m; __syncthreads();
    for (int s = 128; s > 0; s >>= 1) {
        if ((int)threadIdx.x < s) red[threadIdx.x] = fminf(red[threadIdx.x], red[threadIdx.x + s]);
        __syncthreads();
    }
    if (threadIdx.x == 0) out[0] = red[0];
}

// ---------------------------------------------------------------------------
// Detector: brightness/saturation mask
// ---------------------------------------------------------------------------
__global__ void k_mask(const float* __restrict__ src, const float* __restrict__ minval,
                       float* __restrict__ mask, long n) {
    const bool remap = (minval[0] < 0.0f);
    for (long i = (long)blockIdx.x * 256 + threadIdx.x; i < n; i += (long)gridDim.x * 256) {
        long b   = i / HWSZ;
        long pix = i % HWSZ;
        const float* p = src + (size_t)(b * 3) * HWSZ + pix;
        float r  = p[0], g = p[HWSZ], bl = p[2 * (size_t)HWSZ];
        if (remap) { r = (r + 1.0f) * 0.5f; g = (g + 1.0f) * 0.5f; bl = (bl + 1.0f) * 0.5f; }
        float br = 0.299f * r + 0.587f * g + 0.114f * bl;
        float bm = 1.0f / (1.0f + __expf(-20.0f * (br - 0.65f)));
        float mx = fmaxf(r, fmaxf(g, bl));
        float mn = fminf(r, fminf(g, bl));
        float ls = 1.0f / (1.0f + __expf(-20.0f * (0.15f - (mx - mn))));
        mask[i] = bm * ls;
    }
}

// separable 15-tap box filter (count_include_pad => zero-pad, divide by 225 at the end)
__global__ void k_boxh(const float* __restrict__ in, float* __restrict__ out) {
    const long n = (long)BATCH * HWSZ;
    for (long i = (long)blockIdx.x * 256 + threadIdx.x; i < n; i += (long)gridDim.x * 256) {
        long b = i / HWSZ;
        int pix = (int)(i % HWSZ);
        int h = pix / WWID, w = pix % WWID;
        const float* row = in + (size_t)b * HWSZ + (size_t)h * WWID;
        float s = 0.0f;
        #pragma unroll
        for (int d = -7; d <= 7; ++d) {
            int w2 = w + d;
            if (w2 >= 0 && w2 < WWID) s += row[w2];
        }
        out[i] = s;
    }
}

__global__ void k_boxv(const float* __restrict__ in, float* __restrict__ out) {
    const long n = (long)BATCH * HWSZ;
    for (long i = (long)blockIdx.x * 256 + threadIdx.x; i < n; i += (long)gridDim.x * 256) {
        long b = i / HWSZ;
        int pix = (int)(i % HWSZ);
        int h = pix / WWID, w = pix % WWID;
        const float* img = in + (size_t)b * HWSZ;
        float s = 0.0f;
        #pragma unroll
        for (int d = -7; d <= 7; ++d) {
            int h2 = h + d;
            if (h2 >= 0 && h2 < HH) s += img[(size_t)h2 * WWID + w];
        }
        out[i] = s * (1.0f / 225.0f);
    }
}

// ---------------------------------------------------------------------------
// Exact top-k(100) per image: 4-pass radix-select + collect + bitonic sort
// ---------------------------------------------------------------------------
__device__ __forceinline__ unsigned fkey(float x) {
    unsigned u = __float_as_uint(x);
    return (u & 0x80000000u) ? ~u : (u | 0x80000000u);   // monotonic float->uint
}

__global__ __launch_bounds__(256) void k_topk(const float* __restrict__ mask,
                                              const int* __restrict__ rand_sel,
                                              int* __restrict__ coords) {
    __shared__ unsigned hist[256];
    __shared__ unsigned s_prefix, s_rem, s_count;
    __shared__ unsigned long long keys[1024];
    const int tid = threadIdx.x;
    const float* m = mask + (size_t)blockIdx.x * HWSZ;

    unsigned prefix = 0, rem = TOPK_K;
    for (int pass = 0; pass < 4; ++pass) {
        hist[tid] = 0; __syncthreads();
        const int shift = 24 - 8 * pass;
        for (int i = tid; i < HWSZ; i += 256) {
            unsigned k = fkey(m[i]);
            bool ok = (pass == 0) || ((k >> (shift + 8)) == prefix);
            if (ok) atomicAdd(&hist[(k >> shift) & 255], 1u);
        }
        __syncthreads();
        if (tid == 0) {
            unsigned cum = 0; int d = 255;
            for (; d >= 0; --d) { cum += hist[d]; if (cum >= rem) break; }
            if (d < 0) d = 0;
            s_prefix = (prefix << 8) | (unsigned)d;
            s_rem    = rem - (cum - hist[d]);
        }
        __syncthreads();
        prefix = s_prefix; rem = s_rem;
        __syncthreads();
    }
    // prefix == key of the 100th-largest value. Collect all >= it.
    if (tid == 0) s_count = 0;
    __syncthreads();
    for (int i = tid; i < HWSZ; i += 256) {
        unsigned k = fkey(m[i]);
        if (k >= prefix) {
            unsigned slot = atomicAdd(&s_count, 1u);
            if (slot < 1024u)
                keys[slot] = ((unsigned long long)k << 32) | (unsigned)(HWSZ - 1 - i);
        }
    }
    __syncthreads();
    unsigned cnt = s_count < 1024u ? s_count : 1024u;
    for (int i = tid; i < 1024; i += 256)
        if ((unsigned)i >= cnt) keys[i] = 0ull;
    __syncthreads();

    // bitonic sort, descending; ties resolved by larger (HWSZ-1-i) == lower index
    for (unsigned kk = 2; kk <= 1024; kk <<= 1) {
        for (unsigned j = kk >> 1; j > 0; j >>= 1) {
            for (unsigned base = 0; base < 1024; base += 256) {
                unsigned i = base + tid;
                unsigned ixj = i ^ j;
                if (ixj > i) {
                    unsigned long long a = keys[i], b = keys[ixj];
                    bool up = (i & kk) == 0;
                    bool sw = up ? (a < b) : (a > b);
                    if (sw) { keys[i] = b; keys[ixj] = a; }
                }
            }
            __syncthreads();
        }
    }
    if (tid < NPP) {
        int r = rand_sel[blockIdx.x * NPP + tid];
        if (r < 0) r = 0; if (r > 1023) r = 1023;
        unsigned idx = (unsigned)(HWSZ - 1) - (unsigned)(keys[r] & 0xffffffffu);
        int y = (int)(idx / WWID) - PATCH / 2;
        int x = (int)(idx % WWID) - PATCH / 2;
        if (y < 0) y = 0; if (y > HH - PATCH)   y = HH - PATCH;
        if (x < 0) x = 0; if (x > WWID - PATCH) x = WWID - PATCH;
        int p = blockIdx.x * NPP + tid;
        coords[p * 2] = y; coords[p * 2 + 1] = x;
    }
}

// ---------------------------------------------------------------------------
// Weight f32 -> f16 with K padding; layout [M][Kpad], k = c*16 + kh*4 + kw
// ---------------------------------------------------------------------------
__global__ void k_wcvt(const float* __restrict__ in, _Float16* __restrict__ out,
                       int M, int K, int Kp) {
    long total = (long)M * Kp;
    for (long i = (long)blockIdx.x * 256 + threadIdx.x; i < total; i += (long)gridDim.x * 256) {
        int kk = (int)(i % Kp);
        int mr = (int)(i / Kp);
        out[i] = (kk < K) ? (_Float16)in[(size_t)mr * K + kk] : (_Float16)0.0f;
    }
}

// ---------------------------------------------------------------------------
// im2col layer 1: gather patches from pred directly, col layout [N][Kpad=64]
// ---------------------------------------------------------------------------
__global__ void k_im2col1(const float* __restrict__ pred, const int* __restrict__ coords,
                          _Float16* __restrict__ col) {
    const long total = (long)NPATCH * 1024 * 64;
    for (long i = (long)blockIdx.x * 256 + threadIdx.x; i < total; i += (long)gridDim.x * 256) {
        int k = (int)(i & 63);
        long n = i >> 6;
        int p  = (int)(n >> 10);
        int rp = (int)(n & 1023);
        int oy = rp >> 5, ox = rp & 31;
        _Float16 v = (_Float16)0.0f;
        if (k < 48) {
            int c = k >> 4, kh = (k >> 2) & 3, kw = k & 3;
            int iy = oy * 2 - 1 + kh, ix = ox * 2 - 1 + kw;
            if (iy >= 0 && iy < PATCH && ix >= 0 && ix < PATCH) {
                int b  = p >> 2;
                int py = coords[p * 2], px = coords[p * 2 + 1];
                v = (_Float16)pred[((size_t)(b * 3 + c) * HWSZ) + (size_t)(py + iy) * WWID + (px + ix)];
            }
        }
        col[i] = v;
    }
}

// generic im2col: NHWC f16 activations -> col [N][Kpad=Cin*16]
__global__ void k_im2col(const _Float16* __restrict__ act, _Float16* __restrict__ col,
                         int Hin, int Win, int Cin, int Hout, int Wout) {
    const int Kp = Cin * 16;
    const long total = (long)NPATCH * Hout * Wout * Kp;
    for (long i = (long)blockIdx.x * 256 + threadIdx.x; i < total; i += (long)gridDim.x * 256) {
        int k  = (int)(i % Kp);
        long n = i / Kp;
        int img = (int)(n / (Hout * Wout));
        int rp  = (int)(n % (Hout * Wout));
        int oy = rp / Wout, ox = rp % Wout;
        int c = k >> 4, kh = (k >> 2) & 3, kw = k & 3;
        int iy = oy * 2 - 1 + kh, ix = ox * 2 - 1 + kw;
        _Float16 v = (_Float16)0.0f;
        if (iy >= 0 && iy < Hin && ix >= 0 && ix < Win)
            v = act[(((size_t)img * Hin + iy) * Win + ix) * Cin + c];
        col[i] = v;
    }
}

// ---------------------------------------------------------------------------
// WMMA GEMM: C[M,N] = A[M,Kp] * B[N,Kp]^T. One wave owns a 16x64 output tile
// (4 accumulators): the A fragment is loaded once per K-step and reused
// against 4 B fragments -> 5 b128-load-pairs per 4 v_wmma_f32_16x16x32_f16.
// mode 0: bias+leaky -> f16 NHWC activation; mode 1: bias -> f32 raw (BN).
// ---------------------------------------------------------------------------
__global__ __launch_bounds__(256) void k_gemm_wmma(
    const _Float16* __restrict__ A, const _Float16* __restrict__ Bm,
    const float* __restrict__ bias, void* __restrict__ outp,
    int M, int N, int Kp, int mode)
{
    const int wave = threadIdx.x >> 5;
    const int lane = threadIdx.x & 31;
    const int Mt = M >> 4;           // 16-row tiles
    const long Nt = N >> 6;          // 64-col tiles
    const long tile = (long)blockIdx.x * 8 + wave;
    if (tile >= (long)Mt * Nt) return;          // wave-uniform: EXEC stays all-ones
    const int mT = (int)(tile % Mt);
    const int nT = (int)(tile / Mt);
    const int hi  = lane >> 4;                  // 0: lanes 0-15, 1: lanes 16-31
    const int l15 = lane & 15;

    // A fragment rows: lane -> M = l15; 16 contiguous halves at k+16*hi (matches B)
    const _Float16* arow = A + (size_t)(mT * 16 + l15) * Kp + hi * 16;
    // B fragment cols: lane -> N = col; 16 contiguous halves at k (+16 for upper lanes)
    const size_t bstride = (size_t)16 * Kp;
    const _Float16* brow0 = Bm + (size_t)(nT * 64 + l15) * Kp + hi * 16;

    v8f acc0 = {}, acc1 = {}, acc2 = {}, acc3 = {};
    for (int k0 = 0; k0 < Kp; k0 += 32) {
        v8h alo = *(const v8h*)(arow + k0);
        v8h ahi = *(const v8h*)(arow + k0 + 8);
        v16h a = __builtin_shufflevector(alo, ahi, 0,1,2,3,4,5,6,7,8,9,10,11,12,13,14,15);
        v16h b0 = *(const v16h*)(brow0 + k0);
        v16h b1 = *(const v16h*)(brow0 + bstride + k0);
        v16h b2 = *(const v16h*)(brow0 + 2 * bstride + k0);
        v16h b3 = *(const v16h*)(brow0 + 3 * bstride + k0);
        acc0 = __builtin_amdgcn_wmma_f32_16x16x32_f16(false, a, false, b0, (short)0, acc0, false, false);
        acc1 = __builtin_amdgcn_wmma_f32_16x16x32_f16(false, a, false, b1, (short)0, acc1, false, false);
        acc2 = __builtin_amdgcn_wmma_f32_16x16x32_f16(false, a, false, b2, (short)0, acc2, false, false);
        acc3 = __builtin_amdgcn_wmma_f32_16x16x32_f16(false, a, false, b3, (short)0, acc3, false, false);
    }

    // C/D layout: VGPR r -> M = mT*16 + 8*hi + r, N = nT*64 + 16*j + l15
    const int mBase = mT * 16 + hi * 8;
    float bsum[8];
    #pragma unroll
    for (int r = 0; r < 8; ++r) bsum[r] = bias[mBase + r];

    v8f accs[4] = { acc0, acc1, acc2, acc3 };
    #pragma unroll
    for (int j = 0; j < 4; ++j) {
        const int nCol = nT * 64 + j * 16 + l15;
        if (mode == 0) {
            _Float16* o = (_Float16*)outp + (size_t)nCol * M + mBase;
            #pragma unroll
            for (int r = 0; r < 8; ++r) {
                float v = accs[j][r] + bsum[r];
                v = v > 0.0f ? v : 0.2f * v;    // leaky relu (layer 1, no BN)
                o[r] = (_Float16)v;
            }
        } else {
            float* o = (float*)outp + (size_t)nCol * M + mBase;
            #pragma unroll
            for (int r = 0; r < 8; ++r) o[r] = accs[j][r] + bsum[r];
        }
    }
}

// ---------------------------------------------------------------------------
// Training-mode batchnorm: per-channel biased stats, then fused norm+leaky
// ---------------------------------------------------------------------------
__global__ void k_bnstats(const float* __restrict__ raw, int Npos, int C,
                          const float* __restrict__ g, const float* __restrict__ be,
                          float* __restrict__ scale, float* __restrict__ shift) {
    __shared__ float rs[256], rs2[256];
    const int c = blockIdx.x;
    float s = 0.0f, s2 = 0.0f;
    for (int n = threadIdx.x; n < Npos; n += 256) {
        float v = raw[(size_t)n * C + c];
        s += v; s2 += v * v;
    }
    rs[threadIdx.x] = s; rs2[threadIdx.x] = s2; __syncthreads();
    for (int st = 128; st > 0; st >>= 1) {
        if ((int)threadIdx.x < st) {
            rs[threadIdx.x]  += rs[threadIdx.x + st];
            rs2[threadIdx.x] += rs2[threadIdx.x + st];
        }
        __syncthreads();
    }
    if (threadIdx.x == 0) {
        float inv = 1.0f / (float)Npos;
        float mean = rs[0] * inv;
        float var  = rs2[0] * inv - mean * mean;
        float sc = g[c] * rsqrtf(var + 1e-5f);
        scale[c] = sc;
        shift[c] = be[c] - mean * sc;
    }
}

__global__ void k_bnapply(const float* __restrict__ raw, const float* __restrict__ scale,
                          const float* __restrict__ shift, _Float16* __restrict__ act,
                          long n, int C) {
    for (long i = (long)blockIdx.x * 256 + threadIdx.x; i < n; i += (long)gridDim.x * 256) {
        int c = (int)(i % C);
        float v = raw[i] * scale[c] + shift[c];
        v = v > 0.0f ? v : 0.2f * v;
        act[i] = (_Float16)v;
    }
}

// ---------------------------------------------------------------------------
// Conv5 (Cout=1, 4x4, no pad) as a dot product per patch + final loss
// ---------------------------------------------------------------------------
__global__ void k_conv5(const _Float16* __restrict__ act4, const float* __restrict__ w5,
                        const float* __restrict__ b5, float* __restrict__ logits) {
    __shared__ float red[256];
    const int p = blockIdx.x;
    float s = 0.0f;
    for (int k = threadIdx.x; k < 8192; k += 256) {
        int c = k >> 4, kh = (k >> 2) & 3, kw = k & 3;
        s += (float)act4[((size_t)(p * 16) + kh * 4 + kw) * 512 + c] * w5[k];
    }
    red[threadIdx.x] = s; __syncthreads();
    for (int st = 128; st > 0; st >>= 1) {
        if ((int)threadIdx.x < st) red[threadIdx.x] += red[threadIdx.x + st];
        __syncthreads();
    }
    if (threadIdx.x == 0) logits[p] = red[0] + b5[0];
}

__global__ void k_loss(const float* __restrict__ logits, float* __restrict__ out) {
    __shared__ float red[64];
    const int t = threadIdx.x;
    float z = -logits[t];
    float sp = z > 0.0f ? z + log1pf(__expf(-z)) : log1pf(__expf(z));
    red[t] = sp; __syncthreads();
    for (int s = 32; s > 0; s >>= 1) {
        if (t < s) red[t] += red[t + s];
        __syncthreads();
    }
    if (t == 0) out[0] = red[0] * (1.0f / 64.0f);
}

// ---------------------------------------------------------------------------
// Host orchestration
// ---------------------------------------------------------------------------
static inline void launch_gemm(const _Float16* A, const _Float16* B, const float* bias,
                               void* out, int M, int N, int Kp, int mode, hipStream_t s) {
    long tiles = (long)(M / 16) * (N / 64);      // 16x64 tile per wave
    int blocks = (int)((tiles + 7) / 8);
    k_gemm_wmma<<<blocks, 256, 0, s>>>(A, B, bias, out, M, N, Kp, mode);
}

extern "C" void kernel_launch(void* const* d_in, const int* in_sizes, int n_in,
                              void* d_out, int out_size, void* d_ws, size_t ws_size,
                              hipStream_t stream) {
    (void)in_sizes; (void)n_in; (void)out_size; (void)ws_size;
    const float* pred = (const float*)d_in[0];
    const float* src  = (const float*)d_in[1];
    const int*   rsel = (const int*)d_in[2];
    const float *w1 = (const float*)d_in[3],  *b1 = (const float*)d_in[4];
    const float *w2 = (const float*)d_in[5],  *b2 = (const float*)d_in[6];
    const float *g2 = (const float*)d_in[7],  *be2 = (const float*)d_in[8];
    const float *w3 = (const float*)d_in[9],  *b3 = (const float*)d_in[10];
    const float *g3 = (const float*)d_in[11], *be3 = (const float*)d_in[12];
    const float *w4 = (const float*)d_in[13], *b4 = (const float*)d_in[14];
    const float *g4 = (const float*)d_in[15], *be4 = (const float*)d_in[16];
    const float *w5 = (const float*)d_in[17], *b5 = (const float*)d_in[18];

    char* ws = (char*)d_ws;
    size_t off = 0;
    auto take = [&](size_t bytes) -> char* {
        char* p = ws + off;
        off = (off + bytes + 255) & ~(size_t)255;
        return p;
    };
    float*    maskA   = (float*)take(37748736);            // 16*768*768 f32
    float*    maskB   = (float*)take(37748736);            // reused as im2col buffer
    _Float16* colbuf  = (_Float16*)maskB;                  // (max im2col = 33.5 MB < 37.7 MB)
    float*    partial = (float*)take(1024 * 4);
    float*    minval  = (float*)take(256);
    int*      coords  = (int*)take(512);
    _Float16* w1f     = (_Float16*)take(64   * 64   * 2);
    _Float16* w2f     = (_Float16*)take(128  * 1024 * 2);
    _Float16* w3f     = (_Float16*)take(256  * 2048 * 2);
    _Float16* w4f     = (_Float16*)take(512  * 4096 * 2);
    _Float16* act1    = (_Float16*)take((size_t)65536 * 64  * 2);
    _Float16* act2    = (_Float16*)take((size_t)16384 * 128 * 2);
    _Float16* act3    = (_Float16*)take((size_t)4096  * 256 * 2);
    _Float16* act4    = (_Float16*)take((size_t)1024  * 512 * 2);
    float*    raw     = (float*)take((size_t)16384 * 128 * 4);   // max of layers 2-4
    float*    bnsc    = (float*)take(512 * 4);
    float*    bnsh    = (float*)take(512 * 4);
    float*    logits  = (float*)take(256);

    // ---- detector ----
    const long nsrc = (long)BATCH * 3 * HWSZ;
    k_min_partial<<<1024, 256, 0, stream>>>(src, nsrc, partial);
    k_min_final<<<1, 256, 0, stream>>>(partial, 1024, minval);
    const long nmask = (long)BATCH * HWSZ;
    const int mblocks = (int)((nmask + 255) / 256);
    k_mask<<<mblocks, 256, 0, stream>>>(src, minval, maskA, nmask);
    k_boxh<<<mblocks, 256, 0, stream>>>(maskA, maskB);
    k_boxv<<<mblocks, 256, 0, stream>>>(maskB, maskA);

    // ---- top-k + patch coordinates ----
    k_topk<<<BATCH, 256, 0, stream>>>(maskA, rsel, coords);

    // ---- weight conversion (f32 -> f16, [M][Kpad]) ----
    k_wcvt<<<16,   256, 0, stream>>>(w1, w1f, 64,  48,   64);
    k_wcvt<<<512,  256, 0, stream>>>(w2, w2f, 128, 1024, 1024);
    k_wcvt<<<2048, 256, 0, stream>>>(w3, w3f, 256, 2048, 2048);
    k_wcvt<<<8192, 256, 0, stream>>>(w4, w4f, 512, 4096, 4096);

    // ---- layer 1: conv 3->64, fused bias+leaky ----
    k_im2col1<<<16384, 256, 0, stream>>>(pred, coords, colbuf);
    launch_gemm(w1f, colbuf, b1, act1, 64, 65536, 64, 0, stream);

    // ---- layer 2: conv 64->128 + BN + leaky ----
    k_im2col<<<65536, 256, 0, stream>>>(act1, colbuf, 32, 32, 64, 16, 16);
    launch_gemm(w2f, colbuf, b2, raw, 128, 16384, 1024, 1, stream);
    k_bnstats<<<128, 256, 0, stream>>>(raw, 16384, 128, g2, be2, bnsc, bnsh);
    k_bnapply<<<8192, 256, 0, stream>>>(raw, bnsc, bnsh, act2, (long)16384 * 128, 128);

    // ---- layer 3: conv 128->256 + BN + leaky ----
    k_im2col<<<32768, 256, 0, stream>>>(act2, colbuf, 16, 16, 128, 8, 8);
    launch_gemm(w3f, colbuf, b3, raw, 256, 4096, 2048, 1, stream);
    k_bnstats<<<256, 256, 0, stream>>>(raw, 4096, 256, g3, be3, bnsc, bnsh);
    k_bnapply<<<4096, 256, 0, stream>>>(raw, bnsc, bnsh, act3, (long)4096 * 256, 256);

    // ---- layer 4: conv 256->512 + BN + leaky ----
    k_im2col<<<16384, 256, 0, stream>>>(act3, colbuf, 8, 8, 256, 4, 4);
    launch_gemm(w4f, colbuf, b4, raw, 512, 1024, 4096, 1, stream);
    k_bnstats<<<512, 256, 0, stream>>>(raw, 1024, 512, g4, be4, bnsc, bnsh);
    k_bnapply<<<2048, 256, 0, stream>>>(raw, bnsc, bnsh, act4, (long)1024 * 512, 512);

    // ---- layer 5 + loss ----
    k_conv5<<<64, 256, 0, stream>>>(act4, w5, b5, logits);
    k_loss<<<1, 64, 0, stream>>>(logits, (float*)d_out);
}
